// AttnBlock_7198365188778
// MI455X (gfx1250) — matrix operand order repaired and checked by reading, not compiled, not run on hardware
//
#include <hip/hip_runtime.h>
#include <hip/hip_bf16.h>

// ---------------- types ----------------
typedef __attribute__((ext_vector_type(16))) _Float16 v16h;
typedef __attribute__((ext_vector_type(8)))  _Float16 v8h;
typedef __attribute__((ext_vector_type(8)))  float    v8f;

#define B_  4
#define C_  256
#define N_  4096   // H*W = 64*64

#define AS1 __attribute__((address_space(1)))
#define AS3 __attribute__((address_space(3)))

// CDNA5 async global->LDS staging (ASYNCcnt-tracked), guarded for portability.
#if defined(__has_builtin)
#if __has_builtin(__builtin_amdgcn_global_load_async_to_lds_b128) && \
    __has_builtin(__builtin_amdgcn_s_wait_asynccnt)
#define HAVE_ASYNC_LDS 1
#endif
#endif

#if defined(HAVE_ASYNC_LDS)
typedef int v4i_vs __attribute__((vector_size(16)));
static __device__ __forceinline__ void async_b128(const void* g, void* l) {
    __builtin_amdgcn_global_load_async_to_lds_b128(
        (AS1 v4i_vs*)(size_t)g,
        (AS3 v4i_vs*)(unsigned)(size_t)l, 0, 0);
}
static __device__ __forceinline__ void async_wait0() {
    __builtin_amdgcn_s_wait_asynccnt(0);
}
#endif

union V16U { v16h v; v8h h[2]; };

// A-operand fragment (16x32 f16): lane<16 holds K=0..15 contiguous, lane>=16
// holds K=16..31.  Caller passes base already offset by (lane>>4)*16.
static __device__ __forceinline__ v16h ld_afrag(const _Float16* p) {
    V16U u;
    u.h[0] = *(const v8h*)(p);
    u.h[1] = *(const v8h*)(p + 8);
    return u.v;
}
// B-operand fragment (32x16 f16): lane<16 holds K=0..15 contiguous, lane>=16
// holds K=16..31.  Caller passes base already offset by (lane>>4)*16.
static __device__ __forceinline__ v16h ld_bfrag(const _Float16* p) {
    V16U u;
    u.h[0] = *(const v8h*)(p);
    u.h[1] = *(const v8h*)(p + 8);
    return u.v;
}

// ---------------- GroupNorm -> f16 tokens (B,N,C) ----------------
__global__ void k_groupnorm(const float* __restrict__ x,
                            const float* __restrict__ gw,
                            const float* __restrict__ gb,
                            _Float16* __restrict__ h16) {
    int blk = blockIdx.x;          // b*128 + g
    int b = blk >> 7, g = blk & 127;
    int c0 = g * 2;
    const float* xb = x + ((size_t)b * C_ + c0) * N_;   // 2 contiguous channels
    int tid = threadIdx.x;
    float s = 0.f, ss = 0.f;
    for (int i = tid; i < 2 * N_; i += 256) {
        float v = xb[i];
        s += v; ss += v * v;
    }
    __shared__ float r1[256], r2[256];
    r1[tid] = s; r2[tid] = ss;
    __syncthreads();
    for (int w = 128; w > 0; w >>= 1) {
        if (tid < w) { r1[tid] += r1[tid + w]; r2[tid] += r2[tid + w]; }
        __syncthreads();
    }
    float mu   = r1[0] * (1.f / (2 * N_));
    float var  = r2[0] * (1.f / (2 * N_)) - mu * mu;
    float rstd = rsqrtf(var + 1e-5f);
    for (int i = tid; i < 2 * N_; i += 256) {
        int c = c0 + (i >> 12);
        int p = i & (N_ - 1);
        float v = (xb[i] - mu) * rstd * gw[c] + gb[c];
        h16[((size_t)b * N_ + p) * C_ + c] = (_Float16)v;
    }
}

// ---------------- weights f32 -> f16 (wq,wk,wv,wo packed) ----------------
__global__ void k_cvt_w(const float* __restrict__ wq, const float* __restrict__ wk,
                        const float* __restrict__ wv, const float* __restrict__ wo,
                        _Float16* __restrict__ out) {
    int idx = blockIdx.x * blockDim.x + threadIdx.x;   // 0 .. 4*65536-1
    int m = idx >> 16, r = idx & 65535;
    const float* s = (m == 0) ? wq : (m == 1) ? wk : (m == 2) ? wv : wo;
    out[idx] = (_Float16)s[r];
}

// ---------------- QKV projection GEMM: D = h @ W^T + bias ----------------
// grid (rowblk=256, colblk=4), 128 threads (4 waves, 16x64 per wave)
__global__ __launch_bounds__(128) void k_gemm_qkv(
        const _Float16* __restrict__ A,    // h16 [B*N][C]
        const _Float16* __restrict__ Wm,   // [C][C] f16
        const float* __restrict__ bias,
        _Float16* __restrict__ out,
        float scale, int vtrans) {
    int wave = threadIdx.x >> 5, lane = threadIdx.x & 31;
    int m0 = blockIdx.x * 64 + wave * 16;
    int n0 = blockIdx.y * 64;
    int lrow = lane & 15, hi = lane >> 4;
    v8f acc[4] = {};
    for (int kc = 0; kc < 8; ++kc) {
        v16h a = ld_afrag(A + (size_t)(m0 + lrow) * C_ + kc * 32 + hi * 16);
#pragma unroll
        for (int t = 0; t < 4; ++t) {
            v16h bf = ld_bfrag(Wm + (size_t)(n0 + t * 16 + lrow) * C_ + kc * 32 + hi * 16);
            acc[t] = __builtin_amdgcn_wmma_f32_16x16x32_f16(
                false, a, false, bf, (short)0, acc[t], false, false);
        }
    }
#pragma unroll
    for (int t = 0; t < 4; ++t) {
        int n = n0 + t * 16 + lrow;
        float bb = bias[n];
#pragma unroll
        for (int r = 0; r < 8; ++r) {
            int m = m0 + r + hi * 8;
            float v = (acc[t][r] + bb) * scale;
            if (vtrans) {   // store V transposed: vt[b][c][token]
                int bb2 = m >> 12, tok = m & (N_ - 1);
                out[((size_t)bb2 * C_ + n) * N_ + tok] = (_Float16)v;
            } else {
                out[(size_t)m * C_ + n] = (_Float16)v;
            }
        }
    }
}

// ---------------- Flash attention: 64 queries/block, 32-key chunks ----------
__global__ __launch_bounds__(128) void k_attn(
        const _Float16* __restrict__ Q,    // [B*N][C]  (pre-scaled by C^-0.5)
        const _Float16* __restrict__ K,    // [B*N][C]
        const _Float16* __restrict__ VT,   // [B][C][N]
        _Float16* __restrict__ O16) {      // [B*N][C]
    __shared__ __align__(16) _Float16 k_s[32][C_];    // 16 KB
    __shared__ __align__(16) _Float16 vt_s[C_][32];   // 16 KB
    __shared__ __align__(16) _Float16 p_s[4][16][32]; //  4 KB
    int qblk = blockIdx.x, b = blockIdx.y;
    int tid = threadIdx.x, wave = tid >> 5, lane = tid & 31;
    int lrow = lane & 15, hi = lane >> 4;

    // Q fragments resident in VGPRs (16 rows x 256 K per wave)
    const _Float16* qrow = Q + ((size_t)b * N_ + qblk * 64 + wave * 16 + lrow) * C_;
    v16h qf[8];
#pragma unroll
    for (int kc = 0; kc < 8; ++kc) qf[kc] = ld_afrag(qrow + kc * 32 + hi * 16);

    v8f acc[16] = {};
    float mI[8], lI[8];
#pragma unroll
    for (int r = 0; r < 8; ++r) { mI[r] = -1e30f; lI[r] = 0.f; }

    for (int j = 0; j < N_ / 32; ++j) {
        int kb0 = j * 32;
        __syncthreads();
#if defined(HAVE_ASYNC_LDS)
        // DMA cache->LDS via ASYNCcnt path; keeps VGPR/VMEM issue slots free
        // for the WMMA stream.
        for (int i = tid; i < 32 * 32; i += 128) {       // K chunk [32][256]
            int r = i >> 5, seg = i & 31;
            async_b128(K + ((size_t)b * N_ + kb0 + r) * C_ + seg * 8,
                       &k_s[r][seg * 8]);
        }
        for (int i = tid; i < 256 * 4; i += 128) {       // V^T chunk [256][32]
            int r = i >> 2, seg = i & 3;
            async_b128(VT + ((size_t)b * C_ + r) * N_ + kb0 + seg * 8,
                       &vt_s[r][seg * 8]);
        }
        async_wait0();
#else
        for (int i = tid; i < 32 * 32; i += 128) {       // K chunk [32][256]
            int r = i >> 5, seg = i & 31;
            *(v8h*)&k_s[r][seg * 8] =
                *(const v8h*)(K + ((size_t)b * N_ + kb0 + r) * C_ + seg * 8);
        }
        for (int i = tid; i < 256 * 4; i += 128) {       // V^T chunk [256][32]
            int r = i >> 2, seg = i & 3;
            *(v8h*)&vt_s[r][seg * 8] =
                *(const v8h*)(VT + ((size_t)b * C_ + r) * N_ + kb0 + seg * 8);
        }
#endif
        if (j + 1 < N_ / 32) {  // prefetch next K chunk (global_prefetch_b8)
            __builtin_prefetch(K + ((size_t)b * N_ + kb0 + 32 + lane) * C_, 0, 1);
        }
        __syncthreads();

        // scores s = q @ k^T  (two 16x16 key sub-tiles)
        v8f s0 = {}, s1 = {};
        for (int kc = 0; kc < 8; ++kc) {
            v16h a  = qf[kc];
            v16h b0 = ld_bfrag(&k_s[lrow][kc * 32 + hi * 16]);
            v16h b1 = ld_bfrag(&k_s[16 + lrow][kc * 32 + hi * 16]);
            s0 = __builtin_amdgcn_wmma_f32_16x16x32_f16(false, a, false, b0, (short)0, s0, false, false);
            s1 = __builtin_amdgcn_wmma_f32_16x16x32_f16(false, a, false, b1, (short)0, s1, false, false);
        }

        // online softmax (row = M lives at fixed vgpr index, N spread over 16 lanes)
#pragma unroll
        for (int r = 0; r < 8; ++r) {
            float t = fmaxf(s0[r], s1[r]);
#pragma unroll
            for (int msk = 1; msk < 16; msk <<= 1)
                t = fmaxf(t, __shfl_xor(t, msk, 32));
            float mNew = fmaxf(mI[r], t);
            float sc = __expf(mI[r] - mNew);
            float p0 = __expf(s0[r] - mNew);
            float p1 = __expf(s1[r] - mNew);
            float rs = p0 + p1;
#pragma unroll
            for (int msk = 1; msk < 16; msk <<= 1)
                rs += __shfl_xor(rs, msk, 32);
            lI[r] = lI[r] * sc + rs;
            mI[r] = mNew;
#pragma unroll
            for (int ct = 0; ct < 16; ++ct) acc[ct][r] *= sc;
            int M = r + hi * 8;
            p_s[wave][M][lrow]      = (_Float16)p0;   // keys 0..15
            p_s[wave][M][16 + lrow] = (_Float16)p1;   // keys 16..31
        }
        __syncthreads();

        // O += P(16x32) @ V(32x256)
        v16h pf = ld_afrag(&p_s[wave][lrow][hi * 16]);
#pragma unroll
        for (int ct = 0; ct < 16; ++ct) {
            v16h vf = ld_bfrag(&vt_s[ct * 16 + lrow][hi * 16]);
            acc[ct] = __builtin_amdgcn_wmma_f32_16x16x32_f16(
                false, pf, false, vf, (short)0, acc[ct], false, false);
        }
    }

    // finalize: divide by row sums, store f16 tokens
#pragma unroll
    for (int r = 0; r < 8; ++r) {
        float inv = 1.f / lI[r];
        int row = qblk * 64 + wave * 16 + r + hi * 8;
#pragma unroll
        for (int ct = 0; ct < 16; ++ct) {
            O16[((size_t)b * N_ + row) * C_ + ct * 16 + lrow] =
                (_Float16)(acc[ct][r] * inv);
        }
    }
}

// ---------------- faithful layout scramble (B,N,C)->(B,HW,C) --------------
// m[b][i][j] = h2[b][(i&15)*256 + j][i>>4]
__global__ void k_scramble(const _Float16* __restrict__ h2,
                           _Float16* __restrict__ m) {
    int e = blockIdx.x * 256 + threadIdx.x;   // 0 .. 4*4096*256-1 (1<<22)
    int j = e & 255;
    int i = (e >> 8) & 4095;
    int b = e >> 20;
    int tok = ((i & 15) << 8) + j;
    int c = i >> 4;
    m[e] = h2[(((size_t)b << 12) + tok) * C_ + c];
}

// ---------------- out projection + residual + final reshape ----------------
__global__ __launch_bounds__(128) void k_gemm_out(
        const _Float16* __restrict__ A,    // m [B*N][C]
        const _Float16* __restrict__ Wm,   // wo16
        const float* __restrict__ bias,
        const float* __restrict__ x,
        float* __restrict__ out) {
    int wave = threadIdx.x >> 5, lane = threadIdx.x & 31;
    int m0 = blockIdx.x * 64 + wave * 16;
    int n0 = blockIdx.y * 64;
    int lrow = lane & 15, hi = lane >> 4;
    v8f acc[4] = {};
    for (int kc = 0; kc < 8; ++kc) {
        v16h a = ld_afrag(A + (size_t)(m0 + lrow) * C_ + kc * 32 + hi * 16);
#pragma unroll
        for (int t = 0; t < 4; ++t) {
            v16h bf = ld_bfrag(Wm + (size_t)(n0 + t * 16 + lrow) * C_ + kc * 32 + hi * 16);
            acc[t] = __builtin_amdgcn_wmma_f32_16x16x32_f16(
                false, a, false, bf, (short)0, acc[t], false, false);
        }
    }
#pragma unroll
    for (int t = 0; t < 4; ++t) {
        int jcol = n0 + t * 16 + lrow;
        float bb = bias[jcol];
#pragma unroll
        for (int r = 0; r < 8; ++r) {
            int mr = m0 + r + hi * 8;
            int b = mr >> 12, n = mr & (N_ - 1);
            int c = n >> 4;
            int hw = ((n & 15) << 8) + jcol;
            size_t oidx = (((size_t)b * C_ + c) * N_) + hw;
            out[oidx] = x[oidx] + acc[t][r] + bb;
        }
    }
}

// ---------------- launch ----------------
extern "C" void kernel_launch(void* const* d_in, const int* in_sizes, int n_in,
                              void* d_out, int out_size, void* d_ws, size_t ws_size,
                              hipStream_t stream) {
    const float* x  = (const float*)d_in[0];
    const float* gw = (const float*)d_in[1];
    const float* gb = (const float*)d_in[2];
    const float* wq = (const float*)d_in[3];
    const float* bq = (const float*)d_in[4];
    const float* wk = (const float*)d_in[5];
    const float* bk = (const float*)d_in[6];
    const float* wv = (const float*)d_in[7];
    const float* bv = (const float*)d_in[8];
    const float* wo = (const float*)d_in[9];
    const float* bo = (const float*)d_in[10];
    float* out = (float*)d_out;

    char* ws = (char*)d_ws;
    const size_t SZ_TOK = (size_t)B_ * N_ * C_ * sizeof(_Float16); // 8 MB
    const size_t SZ_W   = (size_t)4 * C_ * C_ * sizeof(_Float16);  // 512 KB
    _Float16* h16  = (_Float16*)(ws);
    _Float16* w16  = (_Float16*)(ws + SZ_TOK);
    _Float16* q16  = (_Float16*)(ws + SZ_TOK + SZ_W);
    _Float16* k16  = (_Float16*)((char*)q16 + SZ_TOK);
    _Float16* vt16 = (_Float16*)((char*)k16 + SZ_TOK);
    _Float16* h2   = (_Float16*)((char*)vt16 + SZ_TOK);
    _Float16* mscr = h16;   // reuse: h tokens dead after QKV GEMMs

    (void)in_sizes; (void)n_in; (void)out_size; (void)ws_size;

    k_groupnorm<<<B_ * 128, 256, 0, stream>>>(x, gw, gb, h16);
    k_cvt_w<<<(4 * C_ * C_) / 256, 256, 0, stream>>>(wq, wk, wv, wo, w16);

    const float qscale = 0.0625f;   // 256^-0.5
    k_gemm_qkv<<<dim3(256, 4), 128, 0, stream>>>(h16, w16 + 0 * C_ * C_, bq, q16,  qscale, 0);
    k_gemm_qkv<<<dim3(256, 4), 128, 0, stream>>>(h16, w16 + 1 * C_ * C_, bk, k16,  1.0f,   0);
    k_gemm_qkv<<<dim3(256, 4), 128, 0, stream>>>(h16, w16 + 2 * C_ * C_, bv, vt16, 1.0f,   1);

    k_attn<<<dim3(N_ / 64, B_), 128, 0, stream>>>(q16, k16, vt16, h2);

    k_scramble<<<(B_ * N_ * C_) / 256, 256, 0, stream>>>(h2, mscr);
    k_gemm_out<<<dim3(256, 4), 128, 0, stream>>>(mscr, w16 + 3 * C_ * C_, bo, x, out);
}
